// DTI_3255585210686
// MI455X (gfx1250) — matrix operand order repaired and checked by reading, not compiled, not run on hardware
//
#include <hip/hip_runtime.h>
#include <math.h>

// ---------------- problem constants (match reference) ----------------
#define NUM_NODES 100000
#define N_SUB     20000
#define NEDGE     400000
#define RREL      16
#define NBLK      8
#define HDIM      256
#define SI        32           // HDIM / NBLK
#define PAIRS     4096
#define DFC       512          // 2*HDIM

typedef unsigned short ushort_t;
typedef __attribute__((ext_vector_type(16))) __bf16 v16bf;
typedef __attribute__((ext_vector_type(8)))  float  v8f;

union ABFrag { v16bf bf; uint4 q[2]; };

static __device__ __forceinline__ v8f wmma_bf16(const ABFrag& a, const ABFrag& b, v8f c) {
    // D = A(16x32 bf16) * B(32x16 bf16) + C(16x16 f32)
    return __builtin_amdgcn_wmma_f32_16x16x32_bf16(
        /*neg_a=*/false, a.bf, /*neg_b=*/false, b.bf,
        /*c_mod=*/(short)0, c, /*reuse_a=*/false, /*reuse_b=*/false);
}

static __device__ __forceinline__ unsigned short f2bf(float f) {
    union { float f; unsigned u; } a; a.f = f;
    return (unsigned short)((a.u + 0x7FFFu + ((a.u >> 16) & 1u)) >> 16);
}
static __device__ __forceinline__ unsigned pk2bf(float lo, float hi) {
    return (unsigned)f2bf(lo) | ((unsigned)f2bf(hi) << 16);
}

// CDNA5 packed bf16 atomic add (keeps the 164MB agg buffer bf16 -> L2 resident)
static __device__ __forceinline__ void atom_pk_add_bf16(void* addr, unsigned val) {
    asm volatile("global_atomic_pk_add_bf16 %0, %1, off"
                 :: "v"(addr), "v"(val) : "memory");
}

// ---------------- gather: h0 = emb[node_ids] ----------------
__global__ __launch_bounds__(256) void gather_rows_k(
    const float* __restrict__ emb, const int* __restrict__ ids,
    float* __restrict__ out, int nrows) {
    int t = blockIdx.x * 256 + threadIdx.x;       // one float4 per thread
    int row = t >> 6, j = t & 63;                 // 64 float4 per 256-f row
    if (row >= nrows) return;
    ((float4*)out)[(size_t)row * 64 + j] =
        ((const float4*)emb)[(size_t)ids[row] * 64 + j];
}

// ---------------- bulk f32 -> bf16 (8 elems / thread) ----------------
__global__ __launch_bounds__(256) void cvt_bf16_k(
    const float* __restrict__ in, ushort_t* __restrict__ out, int n8) {
    int t = blockIdx.x * 256 + threadIdx.x;
    if (t >= n8) return;
    const float4* p = (const float4*)in + (size_t)t * 2;
    float4 a = p[0], b = p[1];
    uint4 o;
    o.x = pk2bf(a.x, a.y); o.y = pk2bf(a.z, a.w);
    o.z = pk2bf(b.x, b.y); o.w = pk2bf(b.z, b.w);
    ((uint4*)out)[t] = o;
}

// ---------------- pre-shuffle B matrices into per-lane WMMA fragments ----
// B-frag element j (0..15) of lane L = B[K0 + (L<16?0:8) + (j<8?j:j+8)][colTile*16 + (L&15)]
// stored contiguously: 16 bf16 = 32B per lane -> 2x uint4 loads in GEMM loop.

// BDD weights: B_b[k=r*32+i][o] = w[r][b][i][o];  frag layout [b][ks16][ct2][lane][16]
__global__ __launch_bounds__(256) void prefrag_w_k(
    const float* __restrict__ w, ushort_t* __restrict__ frag) {
    int t = blockIdx.x * 256 + threadIdx.x;     // 131072 threads
    int j = t & 15, lane = (t >> 4) & 31, ct = (t >> 9) & 1;
    int ks = (t >> 10) & 15, b = (t >> 14) & 7;
    int K = ks * 32 + ((lane < 16) ? 0 : 8) + ((j < 8) ? j : j + 8);
    int o = ct * 16 + (lane & 15);
    int r = K >> 5, i = K & 31;
    frag[t] = f2bf(w[(((size_t)r * NBLK + b) * SI + i) * SI + o]);
}

// self-loop: B[k][o] = loop_w[k*256+o]; frag layout [ks8][ct16][lane][16]
__global__ __launch_bounds__(256) void prefrag_lw_k(
    const float* __restrict__ lw, ushort_t* __restrict__ frag) {
    int t = blockIdx.x * 256 + threadIdx.x;     // 65536 threads
    int j = t & 15, lane = (t >> 4) & 31, ct = (t >> 9) & 15, ks = (t >> 13) & 7;
    int K = ks * 32 + ((lane < 16) ? 0 : 8) + ((j < 8) ? j : j + 8);
    int o = ct * 16 + (lane & 15);
    frag[t] = f2bf(lw[(size_t)K * HDIM + o]);
}

// fc1: B[k][o] = fc1_W[k*512+o]; frag layout [ks16][ct32][lane][16]
__global__ __launch_bounds__(256) void prefrag_fc_k(
    const float* __restrict__ W, ushort_t* __restrict__ frag) {
    int t = blockIdx.x * 256 + threadIdx.x;     // 262144 threads
    int j = t & 15, lane = (t >> 4) & 31, ct = (t >> 9) & 31, ks = (t >> 14) & 15;
    int K = ks * 32 + ((lane < 16) ? 0 : 8) + ((j < 8) ? j : j + 8);
    int o = ct * 16 + (lane & 15);
    frag[t] = f2bf(W[(size_t)K * DFC + o]);
}

// ---------------- edge scatter: agg[dst*R+etype] += x[src]*norm (bf16 pk atomics)
__global__ __launch_bounds__(256) void edge_scatter_k(
    const int* __restrict__ src, const int* __restrict__ dst,
    const int* __restrict__ et, const float* __restrict__ norm,
    const float* __restrict__ x, ushort_t* __restrict__ aggb, int nE) {
    int e = blockIdx.x * 8 + (threadIdx.x >> 5);   // one edge per wave32
    if (e >= nE) return;
    int lane = threadIdx.x & 31;
    int s = src[e], d = dst[e], r = et[e];
    float nw = norm[e];
    const float4* xs = (const float4*)(x + (size_t)s * HDIM) + lane * 2;
    float4 a = xs[0], c = xs[1];
    a.x *= nw; a.y *= nw; a.z *= nw; a.w *= nw;
    c.x *= nw; c.y *= nw; c.z *= nw; c.w *= nw;
    ushort_t* base = aggb + ((size_t)d * RREL + r) * HDIM + lane * 8;
    atom_pk_add_bf16(base + 0, pk2bf(a.x, a.y));
    atom_pk_add_bf16(base + 2, pk2bf(a.z, a.w));
    atom_pk_add_bf16(base + 4, pk2bf(c.x, c.y));
    atom_pk_add_bf16(base + 6, pk2bf(c.z, c.w));
}

// ---------------- self-loop GEMM: out = xb(bf16)[N,256] @ loop_w + bias ---
// 8 waves/WG; wave w -> cols [w*32, w*32+32); 16-node tile per WG.
__global__ __launch_bounds__(256) void selfloop_gemm_k(
    const ushort_t* __restrict__ xb, const ushort_t* __restrict__ lwfrag,
    const float* __restrict__ bias, float* __restrict__ out) {
    int n0 = blockIdx.x * 16;
    int w = threadIdx.x >> 5, lane = threadIdx.x & 31;
    int m  = n0 + (lane & 15);
    int kb = (lane < 16) ? 0 : 8;
    v8f acc0 = {}, acc1 = {};
    for (int ks = 0; ks < 8; ++ks) {
        const ushort_t* arow = xb + (size_t)m * HDIM + ks * 32;
        ABFrag A;
        A.q[0] = *(const uint4*)(arow + kb);
        A.q[1] = *(const uint4*)(arow + kb + 16);
        const uint4* bp0 = (const uint4*)lwfrag + ((size_t)(ks * 16 + w * 2 + 0) * 32 + lane) * 2;
        const uint4* bp1 = (const uint4*)lwfrag + ((size_t)(ks * 16 + w * 2 + 1) * 32 + lane) * 2;
        ABFrag B0; B0.q[0] = bp0[0]; B0.q[1] = bp0[1];
        ABFrag B1; B1.q[0] = bp1[0]; B1.q[1] = bp1[1];
        acc0 = wmma_bf16(A, B0, acc0);
        acc1 = wmma_bf16(A, B1, acc1);
    }
    int rbase = n0 + ((lane >> 4) << 3);           // C layout: lanes>=16 -> M+8
    int c0 = w * 32 + (lane & 15);
    float b0 = bias[c0], b1 = bias[c0 + 16];
#pragma unroll
    for (int v = 0; v < 8; ++v) {
        float* o = out + (size_t)(rbase + v) * HDIM;
        o[c0]      = acc0[v] + b0;
        o[c0 + 16] = acc1[v] + b1;
    }
}

// ---------------- BDD GEMM: out[n, b*32+o] += sum_{r,i} agg[n,r,b,i] W[r,b,i,o]
// wave w == diag block b; A row n: 16 chunks of 32 bf16 at stride 256 in aggb.
__global__ __launch_bounds__(256) void bdd_gemm_k(
    const ushort_t* __restrict__ aggb, const ushort_t* __restrict__ wfrag,
    float* __restrict__ out) {
    int n0 = blockIdx.x * 16;
    int b = threadIdx.x >> 5, lane = threadIdx.x & 31;
    int m  = n0 + (lane & 15);
    int kb = (lane < 16) ? 0 : 8;
    v8f acc0 = {}, acc1 = {};
    for (int ks = 0; ks < 16; ++ks) {              // K = 512 = R*SI
        const ushort_t* arow = aggb + (size_t)m * (RREL * HDIM) + ks * HDIM + b * 32;
        ABFrag A;
        A.q[0] = *(const uint4*)(arow + kb);
        A.q[1] = *(const uint4*)(arow + kb + 16);
        const uint4* bp0 = (const uint4*)wfrag + ((size_t)(((b * 16 + ks) * 2 + 0) * 32) + lane) * 2;
        const uint4* bp1 = (const uint4*)wfrag + ((size_t)(((b * 16 + ks) * 2 + 1) * 32) + lane) * 2;
        ABFrag B0; B0.q[0] = bp0[0]; B0.q[1] = bp0[1];
        ABFrag B1; B1.q[0] = bp1[0]; B1.q[1] = bp1[1];
        acc0 = wmma_bf16(A, B0, acc0);
        acc1 = wmma_bf16(A, B1, acc1);
    }
    int rbase = n0 + ((lane >> 4) << 3);
    int c0 = b * 32 + (lane & 15);
#pragma unroll
    for (int v = 0; v < 8; ++v) {
        float* o = out + (size_t)(rbase + v) * HDIM;
        o[c0]      += acc0[v];
        o[c0 + 16] += acc1[v];
    }
}

// ---------------- pair gather + bf16: xpair[p] = [h2[drug], h2[target]] -----
__global__ __launch_bounds__(256) void pair_gather_k(
    const float* __restrict__ h, const int* __restrict__ di,
    const int* __restrict__ ti, ushort_t* __restrict__ xp) {
    int t = blockIdx.x * 256 + threadIdx.x;        // PAIRS*64 threads
    int p = t >> 6, c = t & 63;                    // 64 x uint4 = 512 bf16 / row
    int node = (c < 32) ? di[p] : ti[p];
    int cc = c & 31;
    const float4* s = (const float4*)(h + (size_t)node * HDIM) + cc * 2;
    float4 a = s[0], b = s[1];
    uint4 o;
    o.x = pk2bf(a.x, a.y); o.y = pk2bf(a.z, a.w);
    o.z = pk2bf(b.x, b.y); o.w = pk2bf(b.z, b.w);
    ((uint4*)xp)[(size_t)p * 64 + c] = o;
}

// ---------------- fc1: hid = relu(xpair @ fc1_W + b) [4096,512] -------------
__global__ __launch_bounds__(256) void fc1_gemm_k(
    const ushort_t* __restrict__ xb, const ushort_t* __restrict__ frag,
    const float* __restrict__ bias, float* __restrict__ hid) {
    int n0 = blockIdx.x * 16;
    int g  = blockIdx.y;                            // 0/1 -> col halves
    int w = threadIdx.x >> 5, lane = threadIdx.x & 31;
    int ct0 = g * 16 + w * 2;                       // coltile 0..31
    int m  = n0 + (lane & 15);
    int kb = (lane < 16) ? 0 : 8;
    v8f acc0 = {}, acc1 = {};
    for (int ks = 0; ks < 16; ++ks) {
        const ushort_t* arow = xb + (size_t)m * DFC + ks * 32;
        ABFrag A;
        A.q[0] = *(const uint4*)(arow + kb);
        A.q[1] = *(const uint4*)(arow + kb + 16);
        const uint4* bp0 = (const uint4*)frag + ((size_t)((ks * 32 + ct0 + 0) * 32) + lane) * 2;
        const uint4* bp1 = (const uint4*)frag + ((size_t)((ks * 32 + ct0 + 1) * 32) + lane) * 2;
        ABFrag B0; B0.q[0] = bp0[0]; B0.q[1] = bp0[1];
        ABFrag B1; B1.q[0] = bp1[0]; B1.q[1] = bp1[1];
        acc0 = wmma_bf16(A, B0, acc0);
        acc1 = wmma_bf16(A, B1, acc1);
    }
    int rbase = n0 + ((lane >> 4) << 3);
    int c0 = ct0 * 16 + (lane & 15);
    float b0 = bias[c0], b1 = bias[c0 + 16];
#pragma unroll
    for (int v = 0; v < 8; ++v) {
        float* o = hid + (size_t)(rbase + v) * DFC;
        float r0 = acc0[v] + b0, r1 = acc1[v] + b1;
        o[c0]      = r0 > 0.f ? r0 : 0.f;
        o[c0 + 16] = r1 > 0.f ? r1 : 0.f;
    }
}

// ---------------- fc2 + sigmoid: out[p] = sigmoid(hid[p] . w + b) -----------
__global__ __launch_bounds__(256) void fc2_sigmoid_k(
    const float* __restrict__ hid, const float* __restrict__ w,
    const float* __restrict__ b, float* __restrict__ out) {
    int row = blockIdx.x * 8 + (threadIdx.x >> 5);  // one pair per wave32
    int lane = threadIdx.x & 31;
    const float* hr = hid + (size_t)row * DFC;
    float s = 0.f;
#pragma unroll
    for (int k = 0; k < 16; ++k)
        s += hr[lane + k * 32] * w[lane + k * 32];
#pragma unroll
    for (int off = 16; off; off >>= 1)
        s += __shfl_xor(s, off, 32);
    if (lane == 0) out[row] = 1.f / (1.f + __expf(-(s + b[0])));
}

// =======================================================================
extern "C" void kernel_launch(void* const* d_in, const int* in_sizes, int n_in,
                              void* d_out, int out_size, void* d_ws, size_t ws_size,
                              hipStream_t stream) {
    (void)in_sizes; (void)n_in; (void)out_size; (void)ws_size;
    const int*   drugs   = (const int*)  d_in[0];
    const int*   targets = (const int*)  d_in[1];
    const int*   nodeids = (const int*)  d_in[2];
    const int*   src     = (const int*)  d_in[3];
    const int*   dst     = (const int*)  d_in[4];
    const int*   etype   = (const int*)  d_in[5];
    const float* norm    = (const float*)d_in[6];
    const float* emb     = (const float*)d_in[7];
    const float* w1      = (const float*)d_in[8];
    const float* loop_w1 = (const float*)d_in[9];
    const float* b1      = (const float*)d_in[10];
    const float* w2      = (const float*)d_in[11];
    const float* loop_w2 = (const float*)d_in[12];
    const float* b2      = (const float*)d_in[13];
    const float* fc1_W   = (const float*)d_in[14];
    const float* fc1_b   = (const float*)d_in[15];
    const float* fc2_W   = (const float*)d_in[16];
    const float* fc2_b   = (const float*)d_in[17];
    float* out = (float*)d_out;

    // ---- workspace carve-up (256B aligned) ----
    char* p = (char*)d_ws;
    auto take = [&](size_t bytes) { char* r = p; p += (bytes + 255) & ~(size_t)255; return r; };
    ushort_t* aggb   = (ushort_t*)take((size_t)N_SUB * RREL * HDIM * 2);   // 164 MB (L2-resident)
    float*    h0     = (float*)   take((size_t)N_SUB * HDIM * 4);
    float*    h1     = (float*)   take((size_t)N_SUB * HDIM * 4);
    float*    h2     = (float*)   take((size_t)N_SUB * HDIM * 4);
    ushort_t* xb     = (ushort_t*)take((size_t)N_SUB * HDIM * 2);          // bf16 layer input
    ushort_t* wfrag1 = (ushort_t*)take(131072 * 2);
    ushort_t* wfrag2 = (ushort_t*)take(131072 * 2);
    ushort_t* lwf1   = (ushort_t*)take(65536 * 2);
    ushort_t* lwf2   = (ushort_t*)take(65536 * 2);
    ushort_t* fcfrag = (ushort_t*)take(262144 * 2);
    ushort_t* xpair  = (ushort_t*)take((size_t)PAIRS * DFC * 2);
    float*    hid    = (float*)   take((size_t)PAIRS * DFC * 4);

    // ---- weight pre-fragmentation (cheap, L2-resident) ----
    prefrag_w_k <<<512, 256, 0, stream>>>(w1, wfrag1);
    prefrag_w_k <<<512, 256, 0, stream>>>(w2, wfrag2);
    prefrag_lw_k<<<256, 256, 0, stream>>>(loop_w1, lwf1);
    prefrag_lw_k<<<256, 256, 0, stream>>>(loop_w2, lwf2);
    prefrag_fc_k<<<1024, 256, 0, stream>>>(fc1_W, fcfrag);

    // ---- embedding gather ----
    gather_rows_k<<<(N_SUB * 64 + 255) / 256, 256, 0, stream>>>(emb, nodeids, h0, N_SUB);

    const int nTiles = N_SUB / 16;           // 1250
    const int n8     = N_SUB * HDIM / 8;     // bulk cvt thread count

    // ---- layer 1 ----
    hipMemsetAsync(aggb, 0, (size_t)N_SUB * RREL * HDIM * 2, stream);
    edge_scatter_k<<<NEDGE / 8, 256, 0, stream>>>(src, dst, etype, norm, h0, aggb, NEDGE);
    cvt_bf16_k    <<<(n8 + 255) / 256, 256, 0, stream>>>(h0, xb, n8);
    selfloop_gemm_k<<<nTiles, 256, 0, stream>>>(xb, lwf1, b1, h1);
    bdd_gemm_k    <<<nTiles, 256, 0, stream>>>(aggb, wfrag1, h1);

    // ---- layer 2 ----
    hipMemsetAsync(aggb, 0, (size_t)N_SUB * RREL * HDIM * 2, stream);
    edge_scatter_k<<<NEDGE / 8, 256, 0, stream>>>(src, dst, etype, norm, h1, aggb, NEDGE);
    cvt_bf16_k    <<<(n8 + 255) / 256, 256, 0, stream>>>(h1, xb, n8);
    selfloop_gemm_k<<<nTiles, 256, 0, stream>>>(xb, lwf2, b2, h2);
    bdd_gemm_k    <<<nTiles, 256, 0, stream>>>(aggb, wfrag2, h2);

    // ---- MLP head ----
    pair_gather_k<<<PAIRS * 64 / 256, 256, 0, stream>>>(h2, drugs, targets, xpair);
    fc1_gemm_k   <<<dim3(PAIRS / 16, 2), 256, 0, stream>>>(xpair, fcfrag, fc1_b, hid);
    fc2_sigmoid_k<<<PAIRS / 8, 256, 0, stream>>>(hid, fc2_W, fc2_b, out);
}
